// MultiHeadAttention_39994735460391
// MI455X (gfx1250) — matrix operand order repaired and checked by reading, not compiled, not run on hardware
//
#include <hip/hip_runtime.h>
#include <hip/hip_bf16.h>

typedef __attribute__((ext_vector_type(16))) _Float16 v16h;
typedef __attribute__((ext_vector_type(8)))  _Float16 v8h;
typedef __attribute__((ext_vector_type(4)))  _Float16 v4h;
typedef __attribute__((ext_vector_type(8)))  float    v8f;

constexpr int BB  = 4;
constexpr int SS  = 2048;
constexpr int DD  = 1024;
constexpr int HHH = 16;
constexpr int DKK = 64;

// ---------------------------------------------------------------------------
// Fragment helpers (CDNA5 wave32 WMMA 16x16x32 f16 layouts, ISA 7.12.2)
// ---------------------------------------------------------------------------
__device__ __forceinline__ v16h make_frag(v8h lo, v8h hi) {
  v16h f;
#pragma unroll
  for (int i = 0; i < 8; ++i) { f[i] = lo[i]; f[i + 8] = hi[i]; }
  return f;
}

// A fragment: 16x32 tile at (row0, k0), row-major src, leading dim ld.
// lane<16: row=lane, K = k0 + {0..7, 16..23}; lane>=16: K += 8.
__device__ __forceinline__ v16h load_afrag(const _Float16* __restrict__ src,
                                           int ld, int row0, int k0, int lane) {
  int r  = row0 + (lane & 15);
  int hh = lane >> 4;
  const _Float16* p = src + (size_t)r * ld + k0 + 8 * hh;
  v8h lo = *(const v8h*)p;          // K = k0 + 8*hh + 0..7
  v8h hi = *(const v8h*)(p + 16);   // K = k0 + 16 + 8*hh + 0..7
  return make_frag(lo, hi);
}

// B fragment: 32x16 tile, element (kk,n) = src[(n0+n)*ld + k0+kk].
// lane gives column n; K = k0 + {0..7, 16..23} + 8*half (matches A).
__device__ __forceinline__ v16h load_bfrag(const _Float16* __restrict__ src,
                                           int ld, int n0, int k0, int lane) {
  int n  = n0 + (lane & 15);
  int hh = lane >> 4;
  const _Float16* p = src + (size_t)n * ld + k0 + 8 * hh;
  v8h lo = *(const v8h*)p;
  v8h hi = *(const v8h*)(p + 16);
  return make_frag(lo, hi);
}

// ---------------------------------------------------------------------------
// f32 -> f16 convert (vectorized, grid-stride)
// ---------------------------------------------------------------------------
__global__ void __launch_bounds__(256) cvt_f32_f16_kernel(
    const float* __restrict__ in, _Float16* __restrict__ out, size_t n4) {
  size_t i = (size_t)blockIdx.x * blockDim.x + threadIdx.x;
  size_t stride = (size_t)gridDim.x * blockDim.x;
  const float4* in4 = (const float4*)in;
  v4h* out4 = (v4h*)out;
  for (; i < n4; i += stride) {
    float4 v = in4[i];
    v4h h; h[0] = (_Float16)v.x; h[1] = (_Float16)v.y;
           h[2] = (_Float16)v.z; h[3] = (_Float16)v.w;
    out4[i] = h;
  }
}

// ---------------------------------------------------------------------------
// GEMM: out[m,n] = sum_k A[m,k] * W[n,k] + bias[n]      (X @ W^T + b)
// mode 0: f16 out, layout [b, head, s, dk]   (Q, K)
// mode 1: f16 out, layout [b, head, dk, s]   (V transposed)
// mode 2: f32 out, row-major [M, N]          (final projection)
// Each wave computes a 32x64 tile (B-frags reused across 2 A-frags:
// 12 b128 loads per 8 WMMAs). Block = 8 waves = 256 rows x 64 cols.
// ---------------------------------------------------------------------------
__global__ void __launch_bounds__(256) gemm_wmma_kernel(
    const _Float16* __restrict__ A, const _Float16* __restrict__ W,
    const float* __restrict__ bias, void* __restrict__ out,
    int M, int N, int K, int mode) {
  int lane = threadIdx.x & 31;
  int wave = threadIdx.x >> 5;
  int nb = blockIdx.x * 64;
  int mb = blockIdx.y * 256 + wave * 32;
  int hh = lane >> 4;
  int cn = lane & 15;

  v8f acc[2][4];
#pragma unroll
  for (int f = 0; f < 4; ++f) {
    float bval = bias[nb + 16 * f + cn];
#pragma unroll
    for (int u = 0; u < 2; ++u)
#pragma unroll
      for (int j = 0; j < 8; ++j) acc[u][f][j] = bval;  // bias in C-init
  }

  for (int k0 = 0; k0 < K; k0 += 32) {
    v16h a0 = load_afrag(A, K, mb,      k0, lane);
    v16h a1 = load_afrag(A, K, mb + 16, k0, lane);
#pragma unroll
    for (int f = 0; f < 4; ++f) {
      v16h b = load_bfrag(W, K, nb + 16 * f, k0, lane);
      acc[0][f] = __builtin_amdgcn_wmma_f32_16x16x32_f16(
          false, a0, false, b, (short)0, acc[0][f], false, false);
      acc[1][f] = __builtin_amdgcn_wmma_f32_16x16x32_f16(
          false, a1, false, b, (short)0, acc[1][f], false, false);
    }
  }

#pragma unroll
  for (int u = 0; u < 2; ++u) {
#pragma unroll
    for (int f = 0; f < 4; ++f) {
#pragma unroll
      for (int j = 0; j < 8; ++j) {
        int m = mb + 16 * u + j + 8 * hh;
        int n = nb + 16 * f + cn;
        if (mode == 2) {
          ((float*)out)[(size_t)m * N + n] = acc[u][f][j];
        } else {
          int b = m / SS, s = m % SS;
          int head = n / DKK, dk = n % DKK;
          size_t idx = (mode == 0)
              ? ((((size_t)b * HHH + head) * SS + s) * DKK + dk)
              : ((((size_t)b * HHH + head) * DKK + dk) * SS + s);
          ((_Float16*)out)[idx] = (_Float16)acc[u][f][j];
        }
      }
    }
  }
}

// ---------------------------------------------------------------------------
// Fused flash attention. One wave per (b, head, 32-row q tile).
// Q,K: [b,h,s,dk] f16.  V: [b,h,dk,s] f16 (transposed so PV B-frags are
// contiguous).  32 keys/iter: K and V fragments reused across both
// q-subtiles -> 16 WMMAs per iteration for 16 b128 K/V loads. Online
// softmax in log2 domain (v_exp_f32), P restaged through LDS.
// ---------------------------------------------------------------------------
__global__ void __launch_bounds__(256) attn_wmma_kernel(
    const _Float16* __restrict__ Q, const _Float16* __restrict__ Kf,
    const _Float16* __restrict__ Vt, const int* __restrict__ mask,
    _Float16* __restrict__ O) {
  __shared__ _Float16 plds[8][32 * 32];   // per-wave 32x32 f16 P tile (16 KB)

  int lane = threadIdx.x & 31;
  int wave = threadIdx.x >> 5;
  int gw   = blockIdx.x * 8 + wave;       // 0 .. B*H*(S/32)-1
  int qt   = gw & (SS / 32 - 1);          // 64 q tiles
  int bh   = gw >> 6;
  int head = bh & (HHH - 1);
  int b    = bh >> 4;
  int q0   = qt * 32;
  int hh   = lane >> 4;
  int cn   = lane & 15;

  const _Float16* Qh = Q  + ((size_t)b * HHH + head) * SS * DKK;
  const _Float16* Kh = Kf + ((size_t)b * HHH + head) * SS * DKK;
  const _Float16* Vh = Vt + ((size_t)b * HHH + head) * DKK * SS;
  const int*      Mb = mask + (size_t)b * SS * SS;

  // Q A-fragments stay resident for the whole key loop.
  v16h aq[2][2];
#pragma unroll
  for (int u = 0; u < 2; ++u) {
    aq[u][0] = load_afrag(Qh, DKK, q0 + 16 * u, 0,  lane);
    aq[u][1] = load_afrag(Qh, DKK, q0 + 16 * u, 32, lane);
  }

  v8f acc[2][4] = {};
  float mrow[2][8], lrow[2][8];
#pragma unroll
  for (int u = 0; u < 2; ++u)
#pragma unroll
    for (int j = 0; j < 8; ++j) { mrow[u][j] = -1e30f; lrow[u][j] = 0.0f; }

  _Float16* P = plds[wave];
  const float SCL = 0.125f * 1.4426950408889634f;  // 1/sqrt(dk) * log2(e)

  for (int kb = 0; kb < SS; kb += 32) {
    if (kb + 32 < SS) {
      __builtin_prefetch(Kh + (size_t)(kb + 32) * DKK, 0, 3);
      __builtin_prefetch(Vh + (kb + 32), 0, 3);
    }

    // ---- K fragments (shared by both q-subtiles) ----
    v16h bk[2][2];
#pragma unroll
    for (int t = 0; t < 2; ++t) {
      bk[t][0] = load_bfrag(Kh, DKK, kb + 16 * t, 0,  lane);
      bk[t][1] = load_bfrag(Kh, DKK, kb + 16 * t, 32, lane);
    }

    // ---- scores: 2 q-subtiles x 2 key subtiles x 2 k-chunks = 8 WMMAs ----
    v8f sc[2][2] = {};
#pragma unroll
    for (int u = 0; u < 2; ++u)
#pragma unroll
      for (int t = 0; t < 2; ++t) {
        sc[u][t] = __builtin_amdgcn_wmma_f32_16x16x32_f16(
            false, aq[u][0], false, bk[t][0], (short)0, sc[u][t], false, false);
        sc[u][t] = __builtin_amdgcn_wmma_f32_16x16x32_f16(
            false, aq[u][1], false, bk[t][1], (short)0, sc[u][t], false, false);
      }

    // ---- scale (log2 domain) + mask; C layout row = j + 8*hh, col = cn ----
#pragma unroll
    for (int u = 0; u < 2; ++u)
#pragma unroll
      for (int t = 0; t < 2; ++t) {
        int key = kb + 16 * t + cn;
#pragma unroll
        for (int j = 0; j < 8; ++j) {
          int q = q0 + 16 * u + j + 8 * hh;
          float s = sc[u][t][j] * SCL;
          sc[u][t][j] = (Mb[(size_t)q * SS + key] == 0) ? -1.0e9f : s;
        }
      }

    // ---- online softmax per q-subtile ----
#pragma unroll
    for (int u = 0; u < 2; ++u)
#pragma unroll
      for (int j = 0; j < 8; ++j) {
        float tmax = fmaxf(sc[u][0][j], sc[u][1][j]);
#pragma unroll
        for (int d = 8; d >= 1; d >>= 1)
          tmax = fmaxf(tmax, __shfl_xor(tmax, d, 16));
        float mnew = fmaxf(mrow[u][j], tmax);
        float resc = exp2f(mrow[u][j] - mnew);
        float p0 = exp2f(sc[u][0][j] - mnew);
        float p1 = exp2f(sc[u][1][j] - mnew);
        sc[u][0][j] = p0; sc[u][1][j] = p1;
        float ps = p0 + p1;
#pragma unroll
        for (int d = 8; d >= 1; d >>= 1)
          ps += __shfl_xor(ps, d, 16);
        lrow[u][j] = lrow[u][j] * resc + ps;
        mrow[u][j] = mnew;
#pragma unroll
        for (int f = 0; f < 4; ++f) acc[u][f][j] *= resc;
      }

    // ---- restage P (C layout -> A layout) through LDS ----
#pragma unroll
    for (int u = 0; u < 2; ++u)
#pragma unroll
      for (int t = 0; t < 2; ++t)
#pragma unroll
        for (int j = 0; j < 8; ++j)
          P[(16 * u + j + 8 * hh) * 32 + 16 * t + cn] = (_Float16)sc[u][t][j];
    asm volatile("s_wait_dscnt 0" ::: "memory");   // LDS in-order per wave

    v16h pa[2];
#pragma unroll
    for (int u = 0; u < 2; ++u) {
      const _Float16* pr = P + (16 * u + cn) * 32 + 8 * hh;
      pa[u] = make_frag(*(const v8h*)pr, *(const v8h*)(pr + 16));
    }

    // ---- PV: 4 dk fragments reused across both q-subtiles = 8 WMMAs ----
#pragma unroll
    for (int f = 0; f < 4; ++f) {
      v16h bv = load_bfrag(Vh, SS, 16 * f, kb, lane);
      acc[0][f] = __builtin_amdgcn_wmma_f32_16x16x32_f16(
          false, pa[0], false, bv, (short)0, acc[0][f], false, false);
      acc[1][f] = __builtin_amdgcn_wmma_f32_16x16x32_f16(
          false, pa[1], false, bv, (short)0, acc[1][f], false, false);
    }
  }

  // ---- normalize and store as f16 [b, s, head, dk] for the O-projection ----
#pragma unroll
  for (int u = 0; u < 2; ++u)
#pragma unroll
    for (int j = 0; j < 8; ++j) {
      float inv = 1.0f / lrow[u][j];
      int s = q0 + 16 * u + j + 8 * hh;
#pragma unroll
      for (int f = 0; f < 4; ++f) {
        int dk = 16 * f + cn;
        O[(((size_t)b * SS + s) * HHH + head) * DKK + dk] =
            (_Float16)(acc[u][f][j] * inv);
      }
    }
}

// ---------------------------------------------------------------------------
// Host-side orchestration
// ---------------------------------------------------------------------------
extern "C" void kernel_launch(void* const* d_in, const int* in_sizes, int n_in,
                              void* d_out, int out_size, void* d_ws, size_t ws_size,
                              hipStream_t stream) {
  const float* query = (const float*)d_in[0];
  const float* key   = (const float*)d_in[1];
  const float* value = (const float*)d_in[2];
  const int*   mask  = (const int*)d_in[3];
  const float* wq = (const float*)d_in[4];
  const float* bq = (const float*)d_in[5];
  const float* wk = (const float*)d_in[6];
  const float* bk = (const float*)d_in[7];
  const float* wv = (const float*)d_in[8];
  const float* bv = (const float*)d_in[9];
  const float* wo = (const float*)d_in[10];
  const float* bo = (const float*)d_in[11];

  const size_t nXD = (size_t)BB * SS * DD;   // 8.4M elems
  const size_t nWW = (size_t)DD * DD;        // 1M elems

  char* ws = (char*)d_ws;
  size_t off = 0;
  auto alloc = [&](size_t bytes) {
    char* p = ws + off;
    off += (bytes + 255) & ~(size_t)255;
    return p;
  };
  _Float16* xq  = (_Float16*)alloc(nXD * 2);
  _Float16* xk  = (_Float16*)alloc(nXD * 2);
  _Float16* xv  = (_Float16*)alloc(nXD * 2);
  _Float16* wqh = (_Float16*)alloc(nWW * 2);
  _Float16* wkh = (_Float16*)alloc(nWW * 2);
  _Float16* wvh = (_Float16*)alloc(nWW * 2);
  _Float16* woh = (_Float16*)alloc(nWW * 2);
  _Float16* Qh  = (_Float16*)alloc(nXD * 2);  // [b,h,s,dk]
  _Float16* Kh  = (_Float16*)alloc(nXD * 2);  // [b,h,s,dk]
  _Float16* Vt  = (_Float16*)alloc(nXD * 2);  // [b,h,dk,s]
  _Float16* Att = (_Float16*)alloc(nXD * 2);  // [b,s,h*dk]

  // 1) convert inputs and weights to f16
  cvt_f32_f16_kernel<<<1024, 256, 0, stream>>>(query, xq, nXD / 4);
  cvt_f32_f16_kernel<<<1024, 256, 0, stream>>>(key,   xk, nXD / 4);
  cvt_f32_f16_kernel<<<1024, 256, 0, stream>>>(value, xv, nXD / 4);
  cvt_f32_f16_kernel<<<256, 256, 0, stream>>>(wq, wqh, nWW / 4);
  cvt_f32_f16_kernel<<<256, 256, 0, stream>>>(wk, wkh, nWW / 4);
  cvt_f32_f16_kernel<<<256, 256, 0, stream>>>(wv, wvh, nWW / 4);
  cvt_f32_f16_kernel<<<256, 256, 0, stream>>>(wo, woh, nWW / 4);

  // 2) QKV projections (bias fused); M = B*S = 8192, N = K = 1024
  const int M = BB * SS;
  dim3 ggrid(DD / 64, M / 256);
  gemm_wmma_kernel<<<ggrid, 256, 0, stream>>>(xq, wqh, bq, Qh, M, DD, DD, 0);
  gemm_wmma_kernel<<<ggrid, 256, 0, stream>>>(xk, wkh, bk, Kh, M, DD, DD, 0);
  gemm_wmma_kernel<<<ggrid, 256, 0, stream>>>(xv, wvh, bv, Vt, M, DD, DD, 1);

  // 3) fused flash attention: B*H*(S/32) waves / 8 per block
  int attn_blocks = (BB * HHH * (SS / 32)) / 8;   // 512
  attn_wmma_kernel<<<attn_blocks, 256, 0, stream>>>(Qh, Kh, Vt, mask, Att);

  // 4) output projection -> fp32 d_out
  gemm_wmma_kernel<<<ggrid, 256, 0, stream>>>(Att, woh, bo, d_out, M, DD, DD, 2);
}
